// ChimeraNet_12996571038027
// MI455X (gfx1250) — matrix operand order repaired and checked by reading, not compiled, not run on hardware
//
#include <hip/hip_runtime.h>
#include <stdint.h>

#define BATCH    8192
#define LATENT   1024
#define KENC     784
#define KENC_PAD 800
#define STEPS    10
#define DT_C     0.2f
#define LN_EPS_C 1e-5f
#define LSTR     1032   // padded LDS row stride in bf16 elems (2064 B, 16B aligned, bank-spread)

typedef __attribute__((ext_vector_type(16))) __bf16 v16bf;
typedef __attribute__((ext_vector_type(8)))  float  v8f;

// ---------- bf16 helpers (RNE) ----------
static __device__ __forceinline__ uint16_t f2bf(float f) {
  uint32_t u = __float_as_uint(f);
  u += 0x7FFFu + ((u >> 16) & 1u);
  return (uint16_t)(u >> 16);
}
static __device__ __forceinline__ float bf2f(uint16_t h) {
  return __uint_as_float(((uint32_t)h) << 16);
}

struct Frag8 { uint32_t w[8]; };
static __device__ __forceinline__ v16bf as_bf(const Frag8& f) {
  union { Frag8 f; v16bf v; } x; x.f = f; return x.v;
}

// A-fragment 16x32 bf16 from row-major A[M][Kp].
// lanes 0-15: M=lane, kb=0; lanes 16-31: same rows, kb=8.
// VGPR 0..3 -> K=kb+0..7 (one b128), VGPR 4..7 -> K=kb+16..23 (one b128).
static __device__ __forceinline__ Frag8 loadA(const uint16_t* __restrict__ A,
                                              size_t rowBase, int Kp, int kk, int lane) {
  int m = lane & 15, kb = (lane >> 4) << 3;
  const uint16_t* p = A + (rowBase + (size_t)m) * (size_t)Kp + kk + kb;
  uint4 a = *(const uint4*)p;
  uint4 b = *(const uint4*)(p + 16);
  Frag8 f;
  f.w[0]=a.x; f.w[1]=a.y; f.w[2]=a.z; f.w[3]=a.w;
  f.w[4]=b.x; f.w[5]=b.y; f.w[6]=b.z; f.w[7]=b.w;
  return f;
}

// B-fragment 32x16 bf16 from N-major Bt[N][Kp] (i.e. B transposed).
// lanes 0-15: N=lane, K=0..7,16..23; lanes 16-31: N=lane-16, K=8..15,24..31.
static __device__ __forceinline__ Frag8 loadB(const uint16_t* __restrict__ Bt,
                                              int col, int Kp, int kk, int lane) {
  int n = lane & 15, kb = (lane >> 4) << 3;
  const uint16_t* p = Bt + (size_t)(col + n) * (size_t)Kp + kk + kb;
  uint4 a = *(const uint4*)p;
  uint4 b = *(const uint4*)(p + 16);
  Frag8 f;
  f.w[0]=a.x; f.w[1]=a.y; f.w[2]=a.z; f.w[3]=a.w;
  f.w[4]=b.x; f.w[5]=b.y; f.w[6]=b.z; f.w[7]=b.w;
  return f;
}

// ---------- prep kernels ----------
__global__ void convert_pad_kernel(const float* __restrict__ src, uint16_t* __restrict__ dst,
                                   int rows, int cols, int colsPad) {
  size_t i = (size_t)blockIdx.x * blockDim.x + threadIdx.x;
  size_t total = (size_t)rows * (size_t)colsPad;
  if (i >= total) return;
  int c = (int)(i % colsPad);
  size_t r = i / colsPad;
  dst[i] = (c < cols) ? f2bf(src[r * (size_t)cols + c]) : (uint16_t)0;
}

__global__ void transpose_bf16_kernel(const float* __restrict__ src, uint16_t* __restrict__ dst,
                                      int dim) {
  size_t i = (size_t)blockIdx.x * blockDim.x + threadIdx.x;
  if (i >= (size_t)dim * dim) return;
  int n = (int)(i % dim);          // coalesced read over n
  int k = (int)(i / dim);
  dst[(size_t)n * dim + k] = f2bf(src[i]);
}

// ---------- generic bf16 WMMA GEMM: out[M,1024] = A[M,KT] @ Bt[1024,KT]^T (+bias) ----------
// block = 256 threads (8 waves); 64 rows per block.
// Wave w: row-group (w&1) -> 32 rows (two 16-row A strips sharing each B fragment),
//         N-quarter (w>>1) -> 256 cols (two 128-col chunks).
template<int KT, bool OUT_BF16>
__global__ __launch_bounds__(256, 1)
void gemm_bf16_kernel(const uint16_t* __restrict__ A, const uint16_t* __restrict__ Bt,
                      const float* __restrict__ bias, void* __restrict__ out) {
  const int tid = threadIdx.x, lane = tid & 31, wave = tid >> 5;
  const int rg = wave & 1, nq = wave >> 1;
  const size_t rowBase = (size_t)blockIdx.x * 64 + (size_t)rg * 32;
  const int n = lane & 15, hi = lane >> 4;

  for (int chunk = 0; chunk < 2; ++chunk) {
    const int cb0 = nq * 256 + chunk * 128;
    v8f acc[2][8];
#pragma unroll
    for (int t = 0; t < 8; ++t) {
      float binit = bias ? bias[cb0 + t * 16 + n] : 0.0f;
#pragma unroll
      for (int v = 0; v < 8; ++v) { acc[0][t][v] = binit; acc[1][t][v] = binit; }
    }
    for (int kk = 0; kk < KT; kk += 32) {
      v16bf a0 = as_bf(loadA(A, rowBase,      KT, kk, lane));
      v16bf a1 = as_bf(loadA(A, rowBase + 16, KT, kk, lane));
#pragma unroll
      for (int t = 0; t < 8; ++t) {
        v16bf bv = as_bf(loadB(Bt, cb0 + t * 16, KT, kk, lane));
        acc[0][t] = __builtin_amdgcn_wmma_f32_16x16x32_bf16(
            false, a0, false, bv, (short)0, acc[0][t], false, false);
        acc[1][t] = __builtin_amdgcn_wmma_f32_16x16x32_bf16(
            false, a1, false, bv, (short)0, acc[1][t], false, false);
      }
    }
#pragma unroll
    for (int mb = 0; mb < 2; ++mb) {
#pragma unroll
      for (int t = 0; t < 8; ++t) {
#pragma unroll
        for (int v = 0; v < 8; ++v) {
          size_t r = rowBase + mb * 16 + v + hi * 8;
          size_t c = (size_t)(cb0 + t * 16 + n);
          if (OUT_BF16) ((uint16_t*)out)[r * LATENT + c] = f2bf(acc[mb][t][v]);
          else          ((float*)out)[r * LATENT + c]    = acc[mb][t][v];
        }
      }
    }
  }
}

// ---------- fused recurrent reservoir + LayerNorm + head ----------
// 64 batch rows per block; h double-buffered in LDS (bf16), never leaves the WGP.
// 4-strip tiling: each wave owns ONE 128-col chunk of W_res (zero B redundancy in the WG)
// and all 64 rows as four 16-row A strips -> each B fragment feeds FOUR WMMAs.
// B fragments are explicitly double-buffered so the next tile's loads are in flight
// while the current 4 WMMAs execute (waits become loadcnt<=1 instead of 0).
__global__ __launch_bounds__(256, 1)
void reservoir_kernel(const uint16_t* __restrict__ WresT, const float* __restrict__ drive_in,
                      const float* __restrict__ ln_g, const float* __restrict__ ln_b,
                      const float* __restrict__ W_out, const float* __restrict__ b_out,
                      float* __restrict__ out) {
  __shared__ uint16_t hbuf[2][64 * LSTR];   // 2 * 129 KB = 258 KB of the 320 KB WGP LDS

  const int tid = threadIdx.x, lane = tid & 31, wave = tid >> 5;
  const int n = lane & 15, hi = lane >> 4;
  const int cb0 = wave * 128;               // this wave's private 128-col chunk
  const size_t rowsBase = (size_t)blockIdx.x * 64;

  // Step 1 (h0 == 0): h = DT * tanh(drive_in)
  for (int i = tid; i < 64 * 1024; i += 256) {
    int r = i >> 10, c = i & 1023;
    float d = drive_in[(rowsBase + r) * LATENT + c];
    hbuf[0][r * LSTR + c] = f2bf(DT_C * tanhf(d));
  }
  __syncthreads();

  int cur = 0;
  for (int step = 1; step < STEPS; ++step) {
    const int nxt = cur ^ 1;

    v8f acc[4][8];   // 4 row-strips x 8 col-tiles = 256 accumulator VGPRs
    // init accumulators with the loop-invariant input drive
#pragma unroll
    for (int ms = 0; ms < 4; ++ms)
#pragma unroll
      for (int t = 0; t < 8; ++t)
#pragma unroll
        for (int v = 0; v < 8; ++v)
          acc[ms][t][v] = drive_in[(rowsBase + ms * 16 + v + hi * 8) * LATENT
                                   + cb0 + t * 16 + n];

    // drive += h @ W_res  (A from LDS, B streamed once per WG from L2-resident WresT,
    // software-pipelined with two B-fragment buffers)
    Frag8 fb[2];
    fb[0] = loadB(WresT, cb0, LATENT, 0, lane);
    for (int kk = 0; kk < LATENT; kk += 32) {
      const int m = lane & 15, kb = (lane >> 4) << 3;
      v16bf av[4];
#pragma unroll
      for (int ms = 0; ms < 4; ++ms) {
        const uint16_t* pa = &hbuf[cur][(ms * 16 + m) * LSTR + kk + kb];
        uint4 a0 = *(const uint4*)pa;
        uint4 a1 = *(const uint4*)(pa + 16);
        Frag8 fa;
        fa.w[0]=a0.x; fa.w[1]=a0.y; fa.w[2]=a0.z; fa.w[3]=a0.w;
        fa.w[4]=a1.x; fa.w[5]=a1.y; fa.w[6]=a1.z; fa.w[7]=a1.w;
        av[ms] = as_bf(fa);
      }
#pragma unroll
      for (int t = 0; t < 8; ++t) {
        // prefetch next B tile (wraps into next k-chunk; final overshoot stays in-bounds)
        const int nt  = (t + 1) & 7;
        const int nkk = (t == 7) ? (kk + 32) : kk;
        fb[(t + 1) & 1] = loadB(WresT, cb0 + nt * 16, LATENT, nkk, lane);
        v16bf bv = as_bf(fb[t & 1]);
#pragma unroll
        for (int ms = 0; ms < 4; ++ms) {
          acc[ms][t] = __builtin_amdgcn_wmma_f32_16x16x32_bf16(
              false, av[ms], false, bv, (short)0, acc[ms][t], false, false);
        }
      }
    }

    // Euler update: h' = h + DT*(-h + tanh(drive)) into the other buffer
#pragma unroll
    for (int ms = 0; ms < 4; ++ms) {
#pragma unroll
      for (int t = 0; t < 8; ++t) {
#pragma unroll
        for (int v = 0; v < 8; ++v) {
          int ml = ms * 16 + v + hi * 8;
          int c  = cb0 + t * 16 + n;
          float hold = bf2f(hbuf[cur][ml * LSTR + c]);
          float hn = hold + DT_C * (-hold + tanhf(acc[ms][t][v]));
          hbuf[nxt][ml * LSTR + c] = f2bf(hn);
        }
      }
    }
    __syncthreads();
    cur = nxt;
  }

  // LayerNorm + 10-way head, one wave per row (wave32 shuffle reductions)
  const uint16_t* hf = hbuf[cur];
  for (int r = wave; r < 64; r += 8) {
    float hv[32], gg[32], bb[32];
    float s = 0.0f, s2 = 0.0f;
#pragma unroll
    for (int i = 0; i < 32; ++i) {
      int c = lane + i * 32;
      float v = bf2f(hf[r * LSTR + c]);
      hv[i] = v; gg[i] = ln_g[c]; bb[i] = ln_b[c];
      s += v; s2 += v * v;
    }
#pragma unroll
    for (int off = 16; off > 0; off >>= 1) {
      s  += __shfl_xor(s, off);
      s2 += __shfl_xor(s2, off);
    }
    float mu = s * (1.0f / LATENT);
    float var = s2 * (1.0f / LATENT) - mu * mu;
    float rstd = rsqrtf(var + LN_EPS_C);
#pragma unroll
    for (int o10 = 0; o10 < 10; ++o10) {
      float o = 0.0f;
#pragma unroll
      for (int i = 0; i < 32; ++i) {
        int c = lane + i * 32;
        float hn = (hv[i] - mu) * rstd * gg[i] + bb[i];
        o += hn * W_out[(size_t)o10 * LATENT + c];
      }
#pragma unroll
      for (int off = 16; off > 0; off >>= 1) o += __shfl_xor(o, off);
      if (lane == 0) out[(rowsBase + r) * 10 + o10] = o + b_out[o10];
    }
  }
}

// ---------- launch ----------
extern "C" void kernel_launch(void* const* d_in, const int* in_sizes, int n_in,
                              void* d_out, int out_size, void* d_ws, size_t ws_size,
                              hipStream_t stream) {
  (void)in_sizes; (void)n_in; (void)out_size; (void)ws_size;
  const float* x     = (const float*)d_in[0];   // [8192,784]
  const float* W_enc = (const float*)d_in[1];   // [1024,784]
  const float* W_res = (const float*)d_in[2];   // [1024,1024]
  const float* W_in  = (const float*)d_in[3];   // [1024,1024]
  const float* bias  = (const float*)d_in[4];   // [1024]
  const float* ln_g  = (const float*)d_in[5];
  const float* ln_b  = (const float*)d_in[6];
  const float* W_out = (const float*)d_in[7];   // [10,1024]
  const float* b_out = (const float*)d_in[8];   // [10]
  float* out = (float*)d_out;                   // [8192,10]

  char* ws = (char*)d_ws;
  uint16_t* xb     = (uint16_t*)ws; ws += (size_t)BATCH  * KENC_PAD * 2;  // 13.1 MB
  uint16_t* Wenc_b = (uint16_t*)ws; ws += (size_t)LATENT * KENC_PAD * 2;  //  1.6 MB
  uint16_t* WinT   = (uint16_t*)ws; ws += (size_t)LATENT * LATENT   * 2;  //  2.1 MB
  uint16_t* WresT  = (uint16_t*)ws; ws += (size_t)LATENT * LATENT   * 2;  //  2.1 MB
  uint16_t* u_in_b = (uint16_t*)ws; ws += (size_t)BATCH  * LATENT   * 2;  // 16.8 MB
  float*    drive  = (float*)ws;    ws += (size_t)BATCH  * LATENT   * 4;  // 33.6 MB

  // 1) precision conversion + layout prep
  {
    size_t tx = (size_t)BATCH * KENC_PAD;
    convert_pad_kernel<<<dim3((tx + 255) / 256), dim3(256), 0, stream>>>(x, xb, BATCH, KENC, KENC_PAD);
    size_t te = (size_t)LATENT * KENC_PAD;
    convert_pad_kernel<<<dim3((te + 255) / 256), dim3(256), 0, stream>>>(W_enc, Wenc_b, LATENT, KENC, KENC_PAD);
    size_t tt = (size_t)LATENT * LATENT;
    transpose_bf16_kernel<<<dim3((tt + 255) / 256), dim3(256), 0, stream>>>(W_in,  WinT,  LATENT);
    transpose_bf16_kernel<<<dim3((tt + 255) / 256), dim3(256), 0, stream>>>(W_res, WresT, LATENT);
  }

  // 2) u_in = x @ W_enc^T  (W_enc row-major is already [N][K]); store bf16
  gemm_bf16_kernel<KENC_PAD, true><<<dim3(BATCH / 64), dim3(256), 0, stream>>>(
      xb, Wenc_b, nullptr, (void*)u_in_b);

  // 3) drive_in = u_in @ W_in + bias; store f32
  gemm_bf16_kernel<LATENT, false><<<dim3(BATCH / 64), dim3(256), 0, stream>>>(
      u_in_b, WinT, bias, (void*)drive);

  // 4) fused 10-step reservoir + LayerNorm + head
  reservoir_kernel<<<dim3(BATCH / 64), dim3(256), 0, stream>>>(
      WresT, drive, ln_g, ln_b, W_out, b_out, out);
}
